// TransformerEncoderLayer_44822278701365
// MI455X (gfx1250) — matrix operand order repaired and checked by reading, not compiled, not run on hardware
//
#include <hip/hip_runtime.h>
#include <hip/hip_bf16.h>

// ---------------------------------------------------------------------------
// Conformer encoder layer for MI455X (gfx1250), wave32 + WMMA f16 (f32 accum),
// f16 operand staging + async global->LDS tile copies, double-buffered GEMM.
// S=1024 B=8 D=512 H=8 HD=64 F=2048 K=31
// ---------------------------------------------------------------------------

typedef __attribute__((ext_vector_type(16))) _Float16 v16h;
typedef __attribute__((ext_vector_type(8)))  float    v8f;

union AF { v16h h; unsigned u[8]; };

__device__ __forceinline__ v8f wmma_f16(v16h a, v16h b, v8f c) {
  return __builtin_amdgcn_wmma_f32_16x16x32_f16(false, a, false, b, (short)0, c,
                                                false, false);
}
__device__ __forceinline__ v8f zero8() { v8f z = {0,0,0,0,0,0,0,0}; return z; }
__device__ __forceinline__ float sigmoidf_(float x) { return 1.0f / (1.0f + __expf(-x)); }

// ---- CDNA5 async global->LDS copy (ASYNCcnt path), guarded w/ fallback ----
#if __has_builtin(__builtin_amdgcn_global_load_async_to_lds_b128)
#define HAVE_ASYNC_LDS 1
#else
#define HAVE_ASYNC_LDS 0
#endif

typedef int v4i_ __attribute__((vector_size(16)));
typedef __attribute__((address_space(1))) v4i_ g_v4i;   // global int4
typedef __attribute__((address_space(3))) v4i_ l_v4i;   // LDS int4

__device__ __forceinline__ void cp16(void* lds, const void* g) {
#if HAVE_ASYNC_LDS
  __builtin_amdgcn_global_load_async_to_lds_b128(
      (g_v4i*)(unsigned long long)g,
      (l_v4i*)(unsigned int)(unsigned long long)lds, 0, 0);
#else
  *(uint4*)lds = *(const uint4*)g;
#endif
}

__device__ __forceinline__ void wait_async0() {
#if __has_builtin(__builtin_amdgcn_s_wait_asynccnt)
  __builtin_amdgcn_s_wait_asynccnt(0);
#else
  asm volatile("s_wait_asynccnt 0x0" ::: "memory");
#endif
}

// ---------------------------------------------------------------------------
// GEMM: C/Ch[M,N] = act(A[M,K] * Bt[K,N] + bias) ; fp32 C (w/ residual) and/or
// f16 Ch (optionally transposed). A f16 row-major, Bt f16 pre-transposed [K][N].
// K % 64 == 0. Block 256 thr = 8 waves; tile 128x128x64, double-buffered LDS.
// ---------------------------------------------------------------------------
__device__ __forceinline__ void load_tiles(
    _Float16 (*As)[72], _Float16 (*Bs)[136],
    const _Float16* __restrict__ A, int lda,
    const _Float16* __restrict__ Bt, int ldbt,
    int bm, int bn, int k0, int M, int N, int tid, bool fastA, bool fastB)
{
  // A tile: 128 x 64 halves, row-major copy
  if (fastA) {
    #pragma unroll
    for (int i = 0; i < 4; ++i) {
      int id = tid + i * 256;              // 1024 uint4
      int r = id >> 3, c = (id & 7) << 3;  // 8 uint4 per row
      cp16(&As[r][c], A + (long long)(bm + r) * lda + k0 + c);
    }
  } else {
    #pragma unroll
    for (int i = 0; i < 4; ++i) {
      int id = tid + i * 256;
      int r = id >> 3, c = (id & 7) << 3;
      int gm = bm + r;
      _Float16* d = &As[r][c];
      #pragma unroll
      for (int e = 0; e < 8; ++e)
        d[e] = (gm < M) ? A[(long long)gm * lda + k0 + c + e] : (_Float16)0.f;
    }
  }
  // B tile: 64 x 128 halves ([k][n]), row-major copy (already transposed)
  if (fastB) {
    #pragma unroll
    for (int i = 0; i < 4; ++i) {
      int id = tid + i * 256;
      int r = id >> 4, c = (id & 15) << 3; // 16 uint4 per row
      cp16(&Bs[r][c], Bt + (long long)(k0 + r) * ldbt + bn + c);
    }
  } else {
    #pragma unroll
    for (int i = 0; i < 4; ++i) {
      int id = tid + i * 256;
      int r = id >> 4, c = (id & 15) << 3;
      _Float16* d = &Bs[r][c];
      #pragma unroll
      for (int e = 0; e < 8; ++e) {
        int gn = bn + c + e;
        d[e] = (gn < N) ? Bt[(long long)(k0 + r) * ldbt + gn] : (_Float16)0.f;
      }
    }
  }
}

__global__ __launch_bounds__(256) void gemm_wmma_kernel(
    const _Float16* __restrict__ A, int lda, long long sA,
    const _Float16* __restrict__ Bt, int ldbt, long long sB,
    float* __restrict__ C, int ldc, long long sC,
    _Float16* __restrict__ Ch, int ldch, long long sCh, int chT,
    const float* __restrict__ bias,
    const float* __restrict__ Res, float resScale,
    int M, int N, int K, int act)
{
  __shared__ _Float16 As[2][128][72];   // [m][k], pad to 72 (16B-aligned rows)
  __shared__ _Float16 Bs[2][64][136];   // [k][n], pad to 136

  const int tid    = threadIdx.x;
  const int lane   = tid & 31;
  const int wid    = tid >> 5;
  const int wm     = wid & 1;
  const int wn     = wid >> 1;
  const int halfId = lane >> 4;
  const int ncol   = lane & 15;
  const int mrow   = lane & 15;

  const int bm = blockIdx.y * 128;
  const int bn = blockIdx.x * 128;
  const _Float16* Ab = A + (long long)blockIdx.z * sA;
  const _Float16* Bb = Bt + (long long)blockIdx.z * sB;
  const long long zC  = (long long)blockIdx.z * sC;
  const long long zCh = (long long)blockIdx.z * sCh;

  const bool fastA = (bm + 128 <= M) && ((lda & 7) == 0);
  const bool fastB = (bn + 128 <= N) && ((ldbt & 7) == 0);

  v8f acc[4][2];
  #pragma unroll
  for (int i = 0; i < 4; ++i)
    #pragma unroll
    for (int j = 0; j < 2; ++j) acc[i][j] = zero8();

  const int nk = K >> 6;
  load_tiles(As[0], Bs[0], Ab, lda, Bb, ldbt, bm, bn, 0, M, N, tid, fastA, fastB);

  for (int kt = 0; kt < nk; ++kt) {
    wait_async0();
    __syncthreads();                     // tile kt resident
    if (kt + 1 < nk)
      load_tiles(As[(kt + 1) & 1], Bs[(kt + 1) & 1], Ab, lda, Bb, ldbt,
                 bm, bn, (kt + 1) << 6, M, N, tid, fastA, fastB);
    const int buf = kt & 1;
    #pragma unroll
    for (int ks = 0; ks < 64; ks += 32) {
      AF bf[2];
      #pragma unroll
      for (int j = 0; j < 2; ++j)
        #pragma unroll
        for (int v = 0; v < 8; ++v) {
          int kk = ks + ((v >> 2) << 4) + halfId * 8 + ((v & 3) << 1);
          bf[j].h[2 * v]     = Bs[buf][kk][wn * 32 + j * 16 + ncol];
          bf[j].h[2 * v + 1] = Bs[buf][kk + 1][wn * 32 + j * 16 + ncol];
        }
      #pragma unroll
      for (int i = 0; i < 4; ++i) {
        AF af;
        int m = wm * 64 + i * 16 + mrow;
        #pragma unroll
        for (int v = 0; v < 8; ++v) {
          int kk = ks + ((v >> 2) << 4) + halfId * 8 + ((v & 3) << 1);
          af.u[v] = *(const unsigned*)&As[buf][m][kk];
        }
        #pragma unroll
        for (int j = 0; j < 2; ++j)
          acc[i][j] = wmma_f16(af.h, bf[j].h, acc[i][j]);
      }
    }
    __syncthreads();                     // compute done, buffer reusable
  }

  #pragma unroll
  for (int i = 0; i < 4; ++i) {
    #pragma unroll
    for (int j = 0; j < 2; ++j) {
      int gn = bn + wn * 32 + j * 16 + ncol;
      if (gn >= N) continue;
      float bv = bias ? bias[gn] : 0.0f;
      #pragma unroll
      for (int v = 0; v < 8; ++v) {
        int gm = bm + wm * 64 + i * 16 + v + 8 * halfId;
        if (gm >= M) continue;
        float x = acc[i][j][v] + bv;
        if (act == 1) x = x * sigmoidf_(x);   // swish
        if (C) {
          long long idx = zC + (long long)gm * ldc + gn;
          float o = Res ? Res[idx] + resScale * x : x;
          C[idx] = o;
        }
        if (Ch) {
          if (chT) Ch[zCh + (long long)gn * ldch + gm] = (_Float16)x;
          else     Ch[zCh + (long long)gm * ldch + gn] = (_Float16)x;
        }
      }
    }
  }
}

// ---------------------------------------------------------------------------
// LayerNorm over D=512 (one wave per row); optional swish; fp32 and/or f16 out.
// ---------------------------------------------------------------------------
__global__ __launch_bounds__(256) void ln_kernel(
    const float* __restrict__ x, const float* __restrict__ g,
    const float* __restrict__ b, float* __restrict__ yf,
    _Float16* __restrict__ yh, int rows, int act)
{
  const int wid = threadIdx.x >> 5, lane = threadIdx.x & 31;
  const int row = blockIdx.x * 8 + wid;
  if (row >= rows) return;
  const float* xr = x + (long long)row * 512;
  float s = 0.f, s2 = 0.f;
  #pragma unroll
  for (int i = 0; i < 16; ++i) { float v = xr[lane + i * 32]; s += v; s2 += v * v; }
  #pragma unroll
  for (int m = 16; m >= 1; m >>= 1) {
    s  += __shfl_xor(s,  m, 32);
    s2 += __shfl_xor(s2, m, 32);
  }
  const float mean = s * (1.0f / 512.0f);
  const float var  = s2 * (1.0f / 512.0f) - mean * mean;
  const float inv  = rsqrtf(var + 1e-5f);
  #pragma unroll
  for (int i = 0; i < 16; ++i) {
    int d = lane + i * 32;
    float v = (xr[d] - mean) * inv * g[d] + b[d];
    if (act == 1) v = v * sigmoidf_(v);
    if (yf) yf[(long long)row * 512 + d] = v;
    if (yh) yh[(long long)row * 512 + d] = (_Float16)v;
  }
}

// ---------------------------------------------------------------------------
// Weight converters: fp32 [N][K] -> f16 transposed [K][N]; plain fp32 -> f16.
// ---------------------------------------------------------------------------
__global__ __launch_bounds__(256) void convert_wt_kernel(
    const float* __restrict__ w, _Float16* __restrict__ wt, int N, int K)
{
  size_t i = (size_t)blockIdx.x * 256 + threadIdx.x;
  if (i >= (size_t)N * K) return;
  int k = (int)(i / N), n = (int)(i % N);
  wt[i] = (_Float16)w[(size_t)n * K + k];
}
__global__ __launch_bounds__(256) void convert_h_kernel(
    const float* __restrict__ x, _Float16* __restrict__ y, size_t n)
{
  size_t i = (size_t)blockIdx.x * 256 + threadIdx.x;
  if (i < n) y[i] = (_Float16)x[i];
}

// ---------------------------------------------------------------------------
// Split QKV (S*B,1536) fp32 into f16 per-head (B,H,S,64): q+bu, q+bv, k, v.
// ---------------------------------------------------------------------------
__global__ __launch_bounds__(256) void prep_qkv_kernel(
    const float* __restrict__ qkv, const float* __restrict__ bu,
    const float* __restrict__ bv, _Float16* __restrict__ qu,
    _Float16* __restrict__ qv, _Float16* __restrict__ kt,
    _Float16* __restrict__ vt)
{
  size_t i = (size_t)blockIdx.x * 256 + threadIdx.x;
  if (i >= (size_t)4194304) return;            // S*B*D
  int dall = i & 511;
  size_t sb = i >> 9;                          // s*B + b
  int b = sb & 7; size_t s = sb >> 3;
  int h = dall >> 6; int d = dall & 63;
  size_t base = sb * 1536;
  size_t o = ((((size_t)b * 8 + h) * 1024) + s) * 64 + d;
  float q = qkv[base + dall];
  qu[o] = (_Float16)(q + bu[dall]);
  qv[o] = (_Float16)(q + bv[dall]);
  kt[o] = (_Float16)qkv[base + 512 + dall];
  vt[o] = (_Float16)qkv[base + 1024 + dall];
}

// ---------------------------------------------------------------------------
// Fused flash attention for one batch b: grid (S/128, H), 8 waves x 16 q-rows.
// logits = (q_u.k^T + bd[s, S-1-s+t]) * HD^-0.5 ; online softmax; O += P*V.
// All operands f16, f32 accumulation; f16 output.
// ---------------------------------------------------------------------------
__global__ __launch_bounds__(256) void attn_kernel(
    const _Float16* __restrict__ qu, const _Float16* __restrict__ kt,
    const _Float16* __restrict__ vt, const _Float16* __restrict__ bd,
    _Float16* __restrict__ outp, int bIdx)
{
  const int S_ = 1024, N2 = 2047;
  __shared__ _Float16 Kls[64][34];       // [d][t]
  __shared__ _Float16 Vls[32][72];       // [t][d] (16B-aligned rows for async)
  __shared__ _Float16 Pls[8][16][34];    // per-wave [m][t]

  const int tid = threadIdx.x, lane = tid & 31, wid = tid >> 5;
  const int halfId = lane >> 4, ncol = lane & 15, mrow = lane & 15;
  const int h = blockIdx.y;
  const _Float16* qb = qu + (long long)(bIdx * 8 + h) * S_ * 64;
  const _Float16* kb = kt + (long long)(bIdx * 8 + h) * S_ * 64;
  const _Float16* vb = vt + (long long)(bIdx * 8 + h) * S_ * 64;
  const _Float16* bh = bd + (long long)h * S_ * N2;

  const int qs0 = blockIdx.x * 128 + wid * 16;

  // Q fragments (16x64, two K=32 steps), A-layout: contiguous u32 pair loads.
  AF qf[2];
  #pragma unroll
  for (int st = 0; st < 2; ++st)
    #pragma unroll
    for (int v = 0; v < 8; ++v) {
      int k = st * 32 + ((v >> 2) << 4) + halfId * 8 + ((v & 3) << 1);
      qf[st].u[v] = *(const unsigned*)(qb + (long long)(qs0 + mrow) * 64 + k);
    }

  v8f o[4];
  #pragma unroll
  for (int jj = 0; jj < 4; ++jj) o[jj] = zero8();
  float runm[8], runs[8];
  #pragma unroll
  for (int v = 0; v < 8; ++v) { runm[v] = -1e30f; runs[v] = 0.f; }

  for (int t0 = 0; t0 < S_; t0 += 32) {
    // V tile: straight async copy. K tile: load + transpose to [d][t].
    {
      int t = tid >> 3, d = (tid & 7) << 3;            // 256 uint4 per tensor
      cp16(&Vls[t][d], vb + (long long)(t0 + t) * 64 + d);
      uint4 kx = *(const uint4*)(kb + (long long)(t0 + t) * 64 + d);
      const _Float16* kh = (const _Float16*)&kx;
      #pragma unroll
      for (int e = 0; e < 8; ++e) Kls[d + e][t] = kh[e];
    }
    wait_async0();
    __syncthreads();

    // ac = Q * K^T over HD=64 (two K=32 WMMA steps), two 16-col tiles
    v8f lg[2]; lg[0] = zero8(); lg[1] = zero8();
    #pragma unroll
    for (int j = 0; j < 2; ++j)
      #pragma unroll
      for (int st = 0; st < 2; ++st) {
        AF bf;
        #pragma unroll
        for (int v = 0; v < 8; ++v) {
          int kk = st * 32 + ((v >> 2) << 4) + halfId * 8 + ((v & 3) << 1);
          bf.h[2 * v]     = Kls[kk][j * 16 + ncol];
          bf.h[2 * v + 1] = Kls[kk + 1][j * 16 + ncol];
        }
        lg[j] = wmma_f16(qf[st].h, bf.h, lg[j]);
      }

    // gathered rel-pos bd (f16), scale, online softmax (row stats per VGPR v)
    #pragma unroll
    for (int v = 0; v < 8; ++v) {
      int s_abs = qs0 + 8 * halfId + v;
      long long ro = (long long)s_abs * N2 + (S_ - 1) - s_abs + t0;
      float l0 = (lg[0][v] + (float)bh[ro + ncol]) * 0.125f;
      float l1 = (lg[1][v] + (float)bh[ro + 16 + ncol]) * 0.125f;
      float mx = fmaxf(l0, l1);
      #pragma unroll
      for (int m = 8; m >= 1; m >>= 1) mx = fmaxf(mx, __shfl_xor(mx, m, 16));
      float nm = fmaxf(runm[v], mx);
      float corr = __expf(runm[v] - nm);
      runm[v] = nm;
      float p0 = __expf(l0 - nm), p1 = __expf(l1 - nm);
      float rs = p0 + p1;
      #pragma unroll
      for (int m = 8; m >= 1; m >>= 1) rs += __shfl_xor(rs, m, 16);
      runs[v] = runs[v] * corr + rs;
      #pragma unroll
      for (int jj = 0; jj < 4; ++jj) o[jj][v] *= corr;
      Pls[wid][v + 8 * halfId][ncol]      = (_Float16)p0;
      Pls[wid][v + 8 * halfId][16 + ncol] = (_Float16)p1;
    }

    // re-read P in A-layout (per-wave LDS, in-order within wave)
    AF pf;
    #pragma unroll
    for (int v = 0; v < 8; ++v) {
      int k = ((v >> 2) << 4) + halfId * 8 + ((v & 3) << 1);
      pf.u[v] = *(const unsigned*)&Pls[wid][mrow][k];
    }
    // O += P (16x32) * V (32x64)
    #pragma unroll
    for (int jj = 0; jj < 4; ++jj) {
      AF bf;
      #pragma unroll
      for (int v = 0; v < 8; ++v) {
        int kk = ((v >> 2) << 4) + halfId * 8 + ((v & 3) << 1);
        bf.h[2 * v]     = Vls[kk][jj * 16 + ncol];
        bf.h[2 * v + 1] = Vls[kk + 1][jj * 16 + ncol];
      }
      o[jj] = wmma_f16(pf.h, bf.h, o[jj]);
    }
    __syncthreads();
  }

  #pragma unroll
  for (int jj = 0; jj < 4; ++jj)
    #pragma unroll
    for (int v = 0; v < 8; ++v) {
      int s_abs = qs0 + 8 * halfId + v;
      int d = jj * 16 + ncol;
      outp[((long long)s_abs * 8 + bIdx) * 512 + h * 64 + d] =
          (_Float16)(o[jj][v] / runs[v]);
    }
}

// ---------------------------------------------------------------------------
// GLU + depthwise conv (fp32 elementwise paths)
// ---------------------------------------------------------------------------
__global__ __launch_bounds__(256) void glu_kernel(const float* __restrict__ y,
                                                  float* __restrict__ z)
{
  size_t i = (size_t)blockIdx.x * 256 + threadIdx.x;
  if (i >= (size_t)4194304) return;
  size_t row = i >> 9; int d = i & 511;
  float a = y[row * 1024 + d];
  float g = y[row * 1024 + 512 + d];
  z[i] = a * sigmoidf_(g);
}

__global__ __launch_bounds__(256) void dwconv_kernel(
    const float* __restrict__ z, const float* __restrict__ w,
    const float* __restrict__ bias, float* __restrict__ out)
{
  size_t i = (size_t)blockIdx.x * 256 + threadIdx.x;
  if (i >= (size_t)4194304) return;
  int d = i & 511;
  size_t sb = i >> 9;
  int b = sb & 7; int s = (int)(sb >> 3);
  float acc = bias[d];
  #pragma unroll
  for (int k = 0; k < 31; ++k) {
    int ss = s + k - 15;
    if (ss >= 0 && ss < 1024)
      acc += z[(((size_t)ss * 8 + b) << 9) + d] * w[d * 31 + k];
  }
  out[i] = acc;
}

// ---------------------------------------------------------------------------
// Host orchestration
// ---------------------------------------------------------------------------
static inline void launch_gemm(hipStream_t st,
    const _Float16* A, int lda, long long sA,
    const _Float16* Bt, int ldbt, long long sB,
    float* C, int ldc, long long sC,
    _Float16* Ch, int ldch, long long sCh, int chT,
    const float* bias, const float* res, float rscale,
    int M, int N, int K, int act, int batch)
{
  dim3 grid((N + 127) / 128, (M + 127) / 128, batch);
  gemm_wmma_kernel<<<grid, 256, 0, st>>>(A, lda, sA, Bt, ldbt, sB, C, ldc, sC,
                                         Ch, ldch, sCh, chT, bias, res, rscale,
                                         M, N, K, act);
}

extern "C" void kernel_launch(void* const* d_in, const int* in_sizes, int n_in,
                              void* d_out, int out_size, void* d_ws, size_t ws_size,
                              hipStream_t stream) {
  (void)in_sizes; (void)n_in; (void)out_size; (void)ws_size;
  const float* src     = (const float*)d_in[0];
  const float* pos_emb = (const float*)d_in[1];
  const float* ffm_w1  = (const float*)d_in[2];
  const float* ffm_b1  = (const float*)d_in[3];
  const float* ffm_w2  = (const float*)d_in[4];
  const float* ffm_b2  = (const float*)d_in[5];
  const float* in_w    = (const float*)d_in[6];
  const float* in_b    = (const float*)d_in[7];
  const float* out_w   = (const float*)d_in[8];
  const float* out_b   = (const float*)d_in[9];
  const float* pos_w   = (const float*)d_in[10];
  const float* bias_u  = (const float*)d_in[11];
  const float* bias_v  = (const float*)d_in[12];
  const float* pw1_w   = (const float*)d_in[13];
  const float* pw1_b   = (const float*)d_in[14];
  const float* dw_w    = (const float*)d_in[15];
  const float* dw_b    = (const float*)d_in[16];
  const float* cnorm_g = (const float*)d_in[17];
  const float* cnorm_b = (const float*)d_in[18];
  const float* pw2_w   = (const float*)d_in[19];
  const float* pw2_b   = (const float*)d_in[20];
  const float* ff_w1   = (const float*)d_in[21];
  const float* ff_b1   = (const float*)d_in[22];
  const float* ff_w2   = (const float*)d_in[23];
  const float* ff_b2   = (const float*)d_in[24];
  const float* g_ffm   = (const float*)d_in[25];
  const float* b_ffm   = (const float*)d_in[26];
  const float* g_mha   = (const float*)d_in[27];
  const float* b_mha   = (const float*)d_in[28];
  const float* g_conv  = (const float*)d_in[29];
  const float* b_conv  = (const float*)d_in[30];
  const float* g_ff    = (const float*)d_in[31];
  const float* b_ff    = (const float*)d_in[32];
  const float* g_fin   = (const float*)d_in[33];
  const float* b_fin   = (const float*)d_in[34];

  const size_t SBD = (size_t)1024 * 8 * 512;   // 4194304
  // fp32 region
  float* X     = (float*)d_ws;                 // (S*B, 512)
  float* QKVf  = X + SBD;                      // (S*B, 1536), reused: pw1 out
  float* ZB    = QKVf + 3 * SBD;               // GLU out
  float* CVf   = ZB + SBD;                     // dwconv out
  // f16 region
  _Float16* XLNh = (_Float16*)(CVf + SBD);
  _Float16* HIDh = XLNh + SBD;                 // (S*B,2048) f16; reused as bd
  _Float16* QUh  = HIDh + 4 * SBD;
  _Float16* QVh  = QUh + SBD;
  _Float16* KTh  = QVh + SBD;
  _Float16* VTh  = KTh + SBD;
  _Float16* ATTh = VTh + SBD;
  _Float16* PEh  = ATTh + SBD;                 // pos_emb f16, 2047*512
  _Float16* PBth = PEh + (size_t)2047 * 512;   // pos proj, transposed [512][2047]
  _Float16* WT   = PBth + (size_t)512 * 2047;  // pre-transposed f16 weights
  _Float16* Wffm1 = WT;                        // [512][2048]
  _Float16* Wffm2 = Wffm1 + 1048576;           // [2048][512]
  _Float16* Win   = Wffm2 + 1048576;           // [512][1536]
  _Float16* Wout  = Win   + 786432;            // [512][512]
  _Float16* Wpos  = Wout  + 262144;            // [512][512]
  _Float16* Wpw1  = Wpos  + 262144;            // [512][1024]
  _Float16* Wpw2  = Wpw1  + 524288;            // [512][512]
  _Float16* Wff1  = Wpw2  + 262144;            // [512][2048]
  _Float16* Wff2  = Wff1  + 1048576;           // [2048][512]
  _Float16* BDh   = HIDh;                      // (H,S,2047) f16, reuse
  float*    PW1f  = QKVf;                      // (S*B,1024) fp32, reuse

  const int M = 8192;              // S*B
  const int EW_GRID = (int)(SBD / 256);
  const int LN_GRID = M / 8;

  // ---- one-time f16 conversions (weights transposed to [K][N]) ----
  #define CVT_WT(W, WTp, Nn, Kk) \
    convert_wt_kernel<<<(((Nn)*(Kk))+255)/256, 256, 0, stream>>>(W, WTp, Nn, Kk)
  CVT_WT(ffm_w1, Wffm1, 2048, 512);
  CVT_WT(ffm_w2, Wffm2, 512, 2048);
  CVT_WT(in_w,   Win,   1536, 512);
  CVT_WT(out_w,  Wout,  512, 512);
  CVT_WT(pos_w,  Wpos,  512, 512);
  CVT_WT(pw1_w,  Wpw1,  1024, 512);
  CVT_WT(pw2_w,  Wpw2,  512, 512);
  CVT_WT(ff_w1,  Wff1,  2048, 512);
  CVT_WT(ff_w2,  Wff2,  512, 2048);
  #undef CVT_WT
  convert_h_kernel<<<((2047 * 512) + 255) / 256, 256, 0, stream>>>(
      pos_emb, PEh, (size_t)2047 * 512);

  // x = src
  (void)hipMemcpyAsync(X, src, SBD * sizeof(float), hipMemcpyDeviceToDevice,
                       stream);

  // ---- macaron FFN: x += 0.5 * W2 swish(W1 ln(x) + b1) + b2 ----
  ln_kernel<<<LN_GRID, 256, 0, stream>>>(X, g_ffm, b_ffm, nullptr, XLNh, M, 0);
  launch_gemm(stream, XLNh, 512, 0, Wffm1, 2048, 0, nullptr, 0, 0,
              HIDh, 2048, 0, 0, ffm_b1, nullptr, 0.f, M, 2048, 512, 1, 1);
  launch_gemm(stream, HIDh, 2048, 0, Wffm2, 512, 0, X, 512, 0,
              nullptr, 0, 0, 0, ffm_b2, X, 0.5f, M, 512, 2048, 0, 1);

  // ---- rel-pos MHA ----
  ln_kernel<<<LN_GRID, 256, 0, stream>>>(X, g_mha, b_mha, nullptr, XLNh, M, 0);
  launch_gemm(stream, XLNh, 512, 0, Win, 1536, 0, QKVf, 1536, 0,
              nullptr, 0, 0, 0, in_b, nullptr, 0.f, M, 1536, 512, 0, 1);
  // p = pos_emb @ pos_w^T, stored transposed f16 [dcol][n] for bd GEMM
  launch_gemm(stream, PEh, 512, 0, Wpos, 512, 0, nullptr, 0, 0,
              PBth, 2047, 0, 1, nullptr, nullptr, 0.f, 2047, 512, 512, 0, 1);
  prep_qkv_kernel<<<EW_GRID, 256, 0, stream>>>(QKVf, bias_u, bias_v,
                                               QUh, QVh, KTh, VTh);
  for (int b = 0; b < 8; ++b) {
    // bd[h,s,n] = (q+bias_v)[b,h,s,:] . p[n, h*64:+64]   (batched over H)
    launch_gemm(stream, QVh + (size_t)b * 8 * 1024 * 64, 64, (long long)1024 * 64,
                PBth, 2047, (long long)64 * 2047,
                nullptr, 0, 0,
                BDh, 2047, (long long)1024 * 2047, 0,
                nullptr, nullptr, 0.f, 1024, 2047, 64, 0, 8);
    attn_kernel<<<dim3(8, 8), 256, 0, stream>>>(QUh, KTh, VTh, BDh, ATTh, b);
  }
  launch_gemm(stream, ATTh, 512, 0, Wout, 512, 0, X, 512, 0,
              nullptr, 0, 0, 0, out_b, X, 1.0f, M, 512, 512, 0, 1);

  // ---- conv module ----
  ln_kernel<<<LN_GRID, 256, 0, stream>>>(X, g_conv, b_conv, nullptr, XLNh, M, 0);
  launch_gemm(stream, XLNh, 512, 0, Wpw1, 1024, 0, PW1f, 1024, 0,
              nullptr, 0, 0, 0, pw1_b, nullptr, 0.f, M, 1024, 512, 0, 1);
  glu_kernel<<<EW_GRID, 256, 0, stream>>>(PW1f, ZB);
  dwconv_kernel<<<EW_GRID, 256, 0, stream>>>(ZB, dw_w, dw_b, CVf);
  ln_kernel<<<LN_GRID, 256, 0, stream>>>(CVf, cnorm_g, cnorm_b, nullptr, XLNh, M, 1);
  launch_gemm(stream, XLNh, 512, 0, Wpw2, 512, 0, X, 512, 0,
              nullptr, 0, 0, 0, pw2_b, X, 1.0f, M, 512, 512, 0, 1);

  // ---- final FFN ----
  ln_kernel<<<LN_GRID, 256, 0, stream>>>(X, g_ff, b_ff, nullptr, XLNh, M, 0);
  launch_gemm(stream, XLNh, 512, 0, Wff1, 2048, 0, nullptr, 0, 0,
              HIDh, 2048, 0, 0, ff_b1, nullptr, 0.f, M, 2048, 512, 1, 1);
  launch_gemm(stream, HIDh, 2048, 0, Wff2, 512, 0, X, 512, 0,
              nullptr, 0, 0, 0, ff_b2, X, 1.0f, M, 512, 2048, 0, 1);

  // ---- final LN -> output ----
  ln_kernel<<<LN_GRID, 256, 0, stream>>>(X, g_fin, b_fin, (float*)d_out, nullptr, M, 0);
}
